// Flow_24429773980225
// MI455X (gfx1250) — matrix operand order repaired and checked by reading, not compiled, not run on hardware
//
#include <hip/hip_runtime.h>
#include <hip/hip_bf16.h>

typedef __attribute__((ext_vector_type(16))) __bf16 v16bf;
typedef __attribute__((ext_vector_type(8)))  __bf16 v8bf;
typedef __attribute__((ext_vector_type(4)))  __bf16 v4bf;
typedef __attribute__((ext_vector_type(8)))  float  v8f;
typedef __attribute__((ext_vector_type(4)))  float  v4f;
typedef __attribute__((ext_vector_type(4)))  unsigned int u32x4;
typedef __attribute__((ext_vector_type(8)))  int    i32x8;
typedef __attribute__((ext_vector_type(4)))  int    i32x4;

#define HDIM 1024
#define BDIM 8
#define NDIM 2048
#define MTOK (BDIM * NDIM)   // 16384 tokens

// ---------------------------------------------------------------------------
// TDM: DMA a 64(N-rows) x 128(K) bf16 tile from global into LDS.
// D# group0: count=1 | lds_addr | global_addr | type=2
// D# group1: data_size=2B, pad 4 DWORDs every 64 DWORDs (272B LDS row stride),
//            tensor_dim0=128, tensor_dim1=64, tile_dim0=128, tile_dim1=64,
//            tensor_dim0_stride=1024 elements.
// ---------------------------------------------------------------------------
__device__ __forceinline__ void tdm_load_64x128(const __bf16* g, unsigned lds_off) {
  unsigned long long ga = (unsigned long long)(const void*)g;
  u32x4 g0;
  g0[0] = 1u;                                             // count=1 (valid user D#)
  g0[1] = lds_off;                                        // lds_addr (bytes)
  g0[2] = (unsigned)(ga & 0xffffffffu);                   // global_addr[31:0]
  g0[3] = (unsigned)((ga >> 32) & 0x01ffffffu) | (2u << 30);  // addr[56:32] | type=2
  i32x8 g1;
  g1[0] = (1 << 16) | (1 << 20) | (5 << 22) | (3 << 25);  // 2B elems, pad 16B/256B
  g1[1] = (128 << 16);                                    // tensor_dim0[15:0]=128
  g1[2] = (64 << 16);                                     // tensor_dim1[15:0]=64
  g1[3] = (128 << 16);                                    // tile_dim0=128
  g1[4] = 64;                                             // tile_dim1=64, tile_dim2=0
  g1[5] = 1024;                                           // tensor_dim0_stride lo
  g1[6] = 0;                                              // stride0 hi | stride1 lo
  g1[7] = 0;                                              // stride1 hi
  i32x4 z4 = {0, 0, 0, 0};
#if __clang_major__ >= 23
  i32x8 z8 = {0, 0, 0, 0, 0, 0, 0, 0};
  __builtin_amdgcn_tensor_load_to_lds(g0, g1, z4, z4, z8, 0);
#else
  __builtin_amdgcn_tensor_load_to_lds(g0, g1, z4, z4, 0);
#endif
}

// A-fragment per ISA 16-bit A 16x32 layout: lane group g holds K chunks
// [8g..8g+7] (elements 0-7) and [16+8g..16+8g+7] (elements 8-15).
__device__ __forceinline__ v16bf load_a_frag(const __bf16* p) {
  v8bf lo = *(const v8bf*)p;         // K = 8g + 0..7
  v8bf hi = *(const v8bf*)(p + 16);  // K = 16 + 8g + 0..7
  return __builtin_shufflevector(lo, hi, 0, 1, 2, 3, 4, 5, 6, 7,
                                         8, 9, 10, 11, 12, 13, 14, 15);
}

// ---------------- fp32 -> bf16 conversion (vector x4) ----------------
__global__ __launch_bounds__(256) void cvt_bf16_kernel(const float* __restrict__ src,
                                                       __bf16* __restrict__ dst, int n4) {
  int i = blockIdx.x * blockDim.x + threadIdx.x;
  if (i >= n4) return;
  v4f x = *(const v4f*)(src + (size_t)i * 4);
  v4bf o;
  o[0] = (__bf16)x[0];
  o[1] = (__bf16)x[1];
  o[2] = (__bf16)x[2];
  o[3] = (__bf16)x[3];
  *(v4bf*)(dst + (size_t)i * 4) = o;
}

// ---------------------------------------------------------------------------
// WMMA bf16 GEMM: C[M,1024] = A[M,1024] @ W[1024,1024]^T + bias
// Weight tiles staged block-wide through LDS by the Tensor Data Mover,
// double-buffered (wave 0 issues TDM, s_wait_tensorcnt + barrier handoff).
// Per wave: 32(M) x 64(N) = 2x4 WMMA tiles; 8 waves/block -> 256x64 block tile.
// DUAL:  C = A0@W0^T + A1@W1^T + bias0 + bias1   (fused WE + WN stage)
// GELU:  exact erf GELU epilogue                  (WO stage)
// ---------------------------------------------------------------------------
template <bool DUAL, bool GELU>
__global__ __launch_bounds__(256) void gemm_wmma_kernel(
    const __bf16* __restrict__ A0, const __bf16* __restrict__ W0,
    const __bf16* __restrict__ A1, const __bf16* __restrict__ W1,
    const float* __restrict__ bias0, const float* __restrict__ bias1,
    __bf16* __restrict__ C) {
  constexpr int NW   = DUAL ? 2 : 1;
  constexpr int KCH  = 128;              // K elements per LDS chunk
  constexpr int NCH  = HDIM / KCH;       // 8 chunks
  constexpr int LROW = KCH + 8;          // 136 elems = 272B padded row (TDM pad)
  __shared__ __align__(16) __bf16 ldsW[NW][2][64][LROW];

  const int lane = threadIdx.x & 31;
  const int wave = threadIdx.x >> 5;
  const int r    = lane & 15;            // row-within-tile this lane feeds
  const int kg   = lane >> 4;            // K lane-group

  const int m_base = blockIdx.y * 256 + wave * 32;
  const int n_base = blockIdx.x * 64;

  const __bf16* a0p[2];
  const __bf16* a1p[2];
#pragma unroll
  for (int mi = 0; mi < 2; mi++) {
    a0p[mi] = A0 + (size_t)(m_base + mi * 16 + r) * HDIM + kg * 8;
    if constexpr (DUAL) a1p[mi] = A1 + (size_t)(m_base + mi * 16 + r) * HDIM + kg * 8;
  }

  v8f acc[2][4] = {};

  if (wave == 0) {
    tdm_load_64x128(W0 + (size_t)n_base * HDIM, (unsigned)(size_t)&ldsW[0][0][0][0]);
    if constexpr (DUAL)
      tdm_load_64x128(W1 + (size_t)n_base * HDIM, (unsigned)(size_t)&ldsW[NW - 1][0][0][0]);
  }

  for (int c = 0; c < NCH; c++) {
    const int buf = c & 1;
    if (wave == 0) {
      if (c + 1 < NCH) {
        tdm_load_64x128(W0 + (size_t)n_base * HDIM + (c + 1) * KCH,
                        (unsigned)(size_t)&ldsW[0][buf ^ 1][0][0]);
        if constexpr (DUAL)
          tdm_load_64x128(W1 + (size_t)n_base * HDIM + (c + 1) * KCH,
                          (unsigned)(size_t)&ldsW[NW - 1][buf ^ 1][0][0]);
        __builtin_amdgcn_s_wait_tensorcnt(DUAL ? 2 : 1);  // chunk c complete
      } else {
        __builtin_amdgcn_s_wait_tensorcnt(0);             // last chunk complete
      }
    }
    __syncthreads();  // chunk c visible to all waves

#pragma unroll
    for (int ks = 0; ks < 4; ks++) {
      const int k0 = c * KCH + ks * 32;   // global K base for A
      const int lk = ks * 32 + kg * 8;    // LDS K offset for B (8g / 16+8g chunks)
      v16bf a[2], b[4];
#pragma unroll
      for (int mi = 0; mi < 2; mi++) a[mi] = load_a_frag(a0p[mi] + k0);
#pragma unroll
      for (int ni = 0; ni < 4; ni++) {
        v8bf lo = *(const v8bf*)&ldsW[0][buf][ni * 16 + r][lk];
        v8bf hi = *(const v8bf*)&ldsW[0][buf][ni * 16 + r][lk + 16];
        b[ni] = __builtin_shufflevector(lo, hi, 0, 1, 2, 3, 4, 5, 6, 7,
                                                8, 9, 10, 11, 12, 13, 14, 15);
      }
#pragma unroll
      for (int mi = 0; mi < 2; mi++)
#pragma unroll
        for (int ni = 0; ni < 4; ni++)
          acc[mi][ni] = __builtin_amdgcn_wmma_f32_16x16x32_bf16(
              false, a[mi], false, b[ni], (short)0, acc[mi][ni], false, false);

      if constexpr (DUAL) {
        v16bf a2[2], b2[4];
#pragma unroll
        for (int mi = 0; mi < 2; mi++) a2[mi] = load_a_frag(a1p[mi] + k0);
#pragma unroll
        for (int ni = 0; ni < 4; ni++) {
          v8bf lo = *(const v8bf*)&ldsW[NW - 1][buf][ni * 16 + r][lk];
          v8bf hi = *(const v8bf*)&ldsW[NW - 1][buf][ni * 16 + r][lk + 16];
          b2[ni] = __builtin_shufflevector(lo, hi, 0, 1, 2, 3, 4, 5, 6, 7,
                                                   8, 9, 10, 11, 12, 13, 14, 15);
        }
#pragma unroll
        for (int mi = 0; mi < 2; mi++)
#pragma unroll
          for (int ni = 0; ni < 4; ni++)
            acc[mi][ni] = __builtin_amdgcn_wmma_f32_16x16x32_bf16(
                false, a2[mi], false, b2[ni], (short)0, acc[mi][ni], false, false);
      }
    }
    __syncthreads();  // all waves done with buffer before TDM overwrites it
  }

  // Epilogue: bias (+DUAL bias) (+GELU), store bf16.
  // C/D layout: lane L, acc element i -> M = i + 8*(L>>4), N = (L&15).
#pragma unroll
  for (int ni = 0; ni < 4; ni++) {
    const int n = n_base + ni * 16 + r;
    float bs = bias0[n];
    if constexpr (DUAL) bs += bias1[n];
#pragma unroll
    for (int mi = 0; mi < 2; mi++) {
#pragma unroll
      for (int i = 0; i < 8; i++) {
        float v = acc[mi][ni][i] + bs;
        if constexpr (GELU) v = 0.5f * v * (1.0f + erff(v * 0.70710678118654752f));
        const int m = m_base + mi * 16 + i + 8 * kg;
        C[(size_t)m * HDIM + n] = (__bf16)v;
      }
    }
  }
}

// ---------------- s[b,n] = dot(Q[b,n,:], K[b,n,:]) / 128 : wave-per-token ----------
__global__ __launch_bounds__(256) void qk_diag_kernel(const __bf16* __restrict__ Q,
                                                      const __bf16* __restrict__ K,
                                                      float* __restrict__ s) {
  const int lane = threadIdx.x & 31;
  const int wave = threadIdx.x >> 5;
  const int tok  = blockIdx.x * 8 + wave;
  const size_t base = (size_t)tok * HDIM;
  float acc = 0.0f;
#pragma unroll
  for (int it = 0; it < 4; it++) {
    const int h = it * 256 + lane * 8;
    v8bf q8 = *(const v8bf*)(Q + base + h);
    v8bf k8 = *(const v8bf*)(K + base + h);
#pragma unroll
    for (int j = 0; j < 8; j++) acc = fmaf((float)q8[j], (float)k8[j], acc);
  }
#pragma unroll
  for (int off = 16; off > 0; off >>= 1) acc += __shfl_down(acc, off);
  if (lane == 0) s[tok] = acc * (1.0f / 128.0f);
}

// ---------------- softmax over batch axis (8 entries per n) ------------------------
__global__ __launch_bounds__(256) void batch_softmax_kernel(const float* __restrict__ s,
                                                            float* __restrict__ dg) {
  const int n = blockIdx.x * blockDim.x + threadIdx.x;
  if (n >= NDIM) return;
  float v[BDIM];
  float mx = -3.4e38f;
#pragma unroll
  for (int b = 0; b < BDIM; b++) { v[b] = s[b * NDIM + n]; mx = fmaxf(mx, v[b]); }
  float sum = 0.0f;
#pragma unroll
  for (int b = 0; b < BDIM; b++) { v[b] = __expf(v[b] - mx); sum += v[b]; }
  const float inv = 1.0f / sum;
#pragma unroll
  for (int b = 0; b < BDIM; b++) dg[b * NDIM + n] = v[b] * inv;
}

// ---------------- av[b,n,:] = diag[b,n] * V[b,n,:] (vector x8) ---------------------
__global__ __launch_bounds__(256) void av_scale_kernel(const __bf16* __restrict__ V,
                                                       const float* __restrict__ dg,
                                                       __bf16* __restrict__ av) {
  const size_t i = (size_t)blockIdx.x * blockDim.x + threadIdx.x;
  const size_t base = i * 8;
  const int tok = (int)(i >> 7);   // 8 elems/thread, HDIM/8 = 128 threads per token
  const float d = dg[tok];
  v8bf x = *(const v8bf*)(V + base);
  v8bf o;
#pragma unroll
  for (int j = 0; j < 8; j++) o[j] = (__bf16)(d * (float)x[j]);
  *(v8bf*)(av + base) = o;
}

// ---------------- out[b,n] = dot(A[b,n,:], WA_w) + WA_b : wave-per-token -----------
__global__ __launch_bounds__(256) void wa_proj_kernel(const __bf16* __restrict__ A,
                                                      const float* __restrict__ w,
                                                      const float* __restrict__ bias,
                                                      float* __restrict__ out) {
  const int lane = threadIdx.x & 31;
  const int wave = threadIdx.x >> 5;
  const int tok  = blockIdx.x * 8 + wave;
  const size_t base = (size_t)tok * HDIM;
  float acc = 0.0f;
#pragma unroll
  for (int it = 0; it < 4; it++) {
    const int h = it * 256 + lane * 8;
    v8bf a8 = *(const v8bf*)(A + base + h);
    v8f  w8 = *(const v8f*)(w + h);
#pragma unroll
    for (int j = 0; j < 8; j++) acc = fmaf((float)a8[j], w8[j], acc);
  }
#pragma unroll
  for (int off = 16; off > 0; off >>= 1) acc += __shfl_down(acc, off);
  if (lane == 0) out[tok] = acc + bias[0];
}

extern "C" void kernel_launch(void* const* d_in, const int* in_sizes, int n_in,
                              void* d_out, int out_size, void* d_ws, size_t ws_size,
                              hipStream_t stream) {
  (void)in_sizes; (void)n_in; (void)out_size; (void)ws_size;

  const float* node_t    = (const float*)d_in[0];
  const float* node_next = (const float*)d_in[1];
  const float* WQ_w = (const float*)d_in[2];  const float* WQ_b = (const float*)d_in[3];
  const float* WK_w = (const float*)d_in[4];  const float* WK_b = (const float*)d_in[5];
  const float* WV_w = (const float*)d_in[6];  const float* WV_b = (const float*)d_in[7];
  const float* WO_w = (const float*)d_in[8];  const float* WO_b = (const float*)d_in[9];
  const float* WE_w = (const float*)d_in[10]; const float* WE_b = (const float*)d_in[11];
  const float* WN_w = (const float*)d_in[12]; const float* WN_b = (const float*)d_in[13];
  const float* WA_w = (const float*)d_in[14]; const float* WA_b = (const float*)d_in[15];
  float* out = (float*)d_out;

  // ---- workspace carve-out (~180 MB) ----
  char* p = (char*)d_ws;
  const size_t SZ_TOK = (size_t)MTOK * HDIM * sizeof(__bf16);  // 33.5 MB
  const size_t SZ_W   = (size_t)HDIM * HDIM * sizeof(__bf16);  //  2  MB
  __bf16* Xt = (__bf16*)p; p += SZ_TOK;
  __bf16* Xn = (__bf16*)p; p += SZ_TOK;
  __bf16* Wq = (__bf16*)p; p += SZ_W;
  __bf16* Wk = (__bf16*)p; p += SZ_W;
  __bf16* Wv = (__bf16*)p; p += SZ_W;
  __bf16* Wo = (__bf16*)p; p += SZ_W;
  __bf16* We = (__bf16*)p; p += SZ_W;
  __bf16* Wn = (__bf16*)p; p += SZ_W;
  __bf16* Qb = (__bf16*)p; p += SZ_TOK;
  __bf16* Kb = (__bf16*)p; p += SZ_TOK;
  __bf16* Vb = (__bf16*)p; p += SZ_TOK;
  float*  sS = (float*)p;  p += (size_t)MTOK * sizeof(float);
  float*  dG = (float*)p;  p += (size_t)MTOK * sizeof(float);
  // buffer reuse (producers strictly before consumers):
  __bf16* AV = Qb;  // av overwrites Q (Q dead after sS)
  __bf16* Hh = Kb;  // gelu hidden overwrites K (K dead after sS)
  __bf16* Ab = Xn;  // A_t overwrites node_next copy (dead after V)

  // ---- fp32 -> bf16 conversions ----
  const int n4_tok = (MTOK * HDIM) / 4;   // 4,194,304
  const int n4_w   = (HDIM * HDIM) / 4;   //   262,144
  cvt_bf16_kernel<<<(n4_tok + 255) / 256, 256, 0, stream>>>(node_t,    Xt, n4_tok);
  cvt_bf16_kernel<<<(n4_tok + 255) / 256, 256, 0, stream>>>(node_next, Xn, n4_tok);
  cvt_bf16_kernel<<<(n4_w + 255) / 256, 256, 0, stream>>>(WQ_w, Wq, n4_w);
  cvt_bf16_kernel<<<(n4_w + 255) / 256, 256, 0, stream>>>(WK_w, Wk, n4_w);
  cvt_bf16_kernel<<<(n4_w + 255) / 256, 256, 0, stream>>>(WV_w, Wv, n4_w);
  cvt_bf16_kernel<<<(n4_w + 255) / 256, 256, 0, stream>>>(WO_w, Wo, n4_w);
  cvt_bf16_kernel<<<(n4_w + 255) / 256, 256, 0, stream>>>(WE_w, We, n4_w);
  cvt_bf16_kernel<<<(n4_w + 255) / 256, 256, 0, stream>>>(WN_w, Wn, n4_w);

  const dim3 ggemm(HDIM / 64, MTOK / 256);  // 16 x 64
  const dim3 blk(256);

  // Q, K, V projections
  gemm_wmma_kernel<false, false><<<ggemm, blk, 0, stream>>>(Xt, Wq, nullptr, nullptr, WQ_b, nullptr, Qb);
  gemm_wmma_kernel<false, false><<<ggemm, blk, 0, stream>>>(Xn, Wk, nullptr, nullptr, WK_b, nullptr, Kb);
  gemm_wmma_kernel<false, false><<<ggemm, blk, 0, stream>>>(Xn, Wv, nullptr, nullptr, WV_b, nullptr, Vb);

  // diagonal attention scores + batch-axis softmax + V scaling
  qk_diag_kernel<<<MTOK / 8, 256, 0, stream>>>(Qb, Kb, sS);
  batch_softmax_kernel<<<(NDIM + 255) / 256, 256, 0, stream>>>(sS, dG);
  av_scale_kernel<<<(MTOK * HDIM / 8) / 256, 256, 0, stream>>>(Vb, dG, AV);

  // h = gelu(av @ WO^T + bo)
  gemm_wmma_kernel<false, true><<<ggemm, blk, 0, stream>>>(AV, Wo, nullptr, nullptr, WO_b, nullptr, Hh);

  // A = node_t @ WE^T + be + h @ WN^T + bn  (fused dual GEMM)
  gemm_wmma_kernel<true, false><<<ggemm, blk, 0, stream>>>(Xt, We, Hh, Wn, WE_b, WN_b, Ab);

  // out = A @ WA_w^T + WA_b
  wa_proj_kernel<<<MTOK / 8, 256, 0, stream>>>(Ab, WA_w, WA_b, out);
}
